// Model_46634754900484
// MI455X (gfx1250) — compile-verified, matrix operand order checked by reading, not run on hardware
//
#include <hip/hip_runtime.h>
#include <hip/hip_bf16.h>

// ---------------------------------------------------------------------------
// MI455X (gfx1250) implementation. All heavy matmuls run on
// v_wmma_f32_16x16x32_bf16 (wave32). fp32 params are converted (and where
// needed transposed) to bf16 N-major weights once per launch. The GEMM
// supports a split-A form A = [A1 | A2] along K with per-part row-tiling
// (row % mod), so the reference's jnp.concatenate/jnp.tile feeds are read
// in place with zero materialized concat buffers. Epilogues fold bias +
// optional tanh and emit bf16 activation copies for the next matmul.
// ---------------------------------------------------------------------------

typedef __bf16 bf16_t;
typedef __attribute__((ext_vector_type(16))) __bf16 v16bf;
typedef __attribute__((ext_vector_type(8)))  float  v8f;

union FragBF { v16bf v; uint4 q[2]; };

__device__ __forceinline__ float sigmoidf_(float x) { return 1.0f / (1.0f + __expf(-x)); }

// One 16x64 wave step: load A fragment at Arow+kA, four B fragments at
// B*+kB, issue 4 WMMAs. Fragment layout per CDNA5 ISA 7.12.2 (16-bit, wave32):
// per lane the 16 bf16 elements are two contiguous 8-element runs at
// k0 + half*8 and k0 + 16 + half*8 -> two b128 loads per fragment.
__device__ __forceinline__ void mma_chunk(
    const bf16_t* __restrict__ Arow, int kA,
    const bf16_t* __restrict__ B0, const bf16_t* __restrict__ B1,
    const bf16_t* __restrict__ B2, const bf16_t* __restrict__ B3, int kB,
    v8f& acc0, v8f& acc1, v8f& acc2, v8f& acc3)
{
    FragBF a, b0, b1, b2, b3;
    a.q[0]  = *(const uint4*)(Arow + kA);
    a.q[1]  = *(const uint4*)(Arow + kA + 16);
    b0.q[0] = *(const uint4*)(B0 + kB);
    b0.q[1] = *(const uint4*)(B0 + kB + 16);
    b1.q[0] = *(const uint4*)(B1 + kB);
    b1.q[1] = *(const uint4*)(B1 + kB + 16);
    b2.q[0] = *(const uint4*)(B2 + kB);
    b2.q[1] = *(const uint4*)(B2 + kB + 16);
    b3.q[0] = *(const uint4*)(B3 + kB);
    b3.q[1] = *(const uint4*)(B3 + kB + 16);
    acc0 = __builtin_amdgcn_wmma_f32_16x16x32_bf16(false, a.v, false, b0.v, (short)0, acc0, false, false);
    acc1 = __builtin_amdgcn_wmma_f32_16x16x32_bf16(false, a.v, false, b1.v, (short)0, acc1, false, false);
    acc2 = __builtin_amdgcn_wmma_f32_16x16x32_bf16(false, a.v, false, b2.v, (short)0, acc2, false, false);
    acc3 = __builtin_amdgcn_wmma_f32_16x16x32_bf16(false, a.v, false, b3.v, (short)0, acc3, false, false);
}

// ---------------------------------------------------------------------------
// GEMM: C(MxN,f32) = act( [A1|A2](MxK,bf16) * Bt(NxK,bf16,ldb)^T + bias[N] )
//   A1 covers k in [0,K1) with leading dim lda1; row index = mod1 ? r%mod1 : r
//   A2 covers k in [K1,K) with leading dim lda2; row index = mod2 ? r%mod2 : r
//   (A2 may be null, then K1 == K). Optional bf16 copy Cb.
// Requires M%128==0, N%64==0, K1%32==0, K%32==0.
// Block: 256 threads (8 waves). Block tile 128x64; each wave: 16x64.
// ---------------------------------------------------------------------------
template<int ACT>  // 0 = none, 1 = tanh
__global__ __launch_bounds__(256)
void gemm_wmma_bf16(const bf16_t* __restrict__ A1, int lda1, int mod1,
                    const bf16_t* __restrict__ A2, int lda2, int mod2,
                    const bf16_t* __restrict__ Bt, int ldb,
                    const float* __restrict__ bias,
                    float* __restrict__ C, bf16_t* __restrict__ Cb,
                    int M, int N, int K1, int K)
{
    const int lane = threadIdx.x & 31;
    const int wid  = threadIdx.x >> 5;
    const int half = lane >> 4;
    const int l16  = lane & 15;
    const int m0 = blockIdx.x * 128 + wid * 16;
    const int n0 = blockIdx.y * 64;
    const int r  = m0 + l16;

    const bf16_t* Arow1 = A1 + (size_t)(mod1 ? (r % mod1) : r) * lda1;
    const bf16_t* Arow2 = A2 ? A2 + (size_t)(mod2 ? (r % mod2) : r) * lda2 : nullptr;
    const bf16_t* B0 = Bt + (size_t)(n0 +  0 + l16) * ldb;
    const bf16_t* B1 = Bt + (size_t)(n0 + 16 + l16) * ldb;
    const bf16_t* B2 = Bt + (size_t)(n0 + 32 + l16) * ldb;
    const bf16_t* B3 = Bt + (size_t)(n0 + 48 + l16) * ldb;

    v8f acc0 = {}, acc1 = {}, acc2 = {}, acc3 = {};

    for (int k0 = 0; k0 < K1; k0 += 32)
        mma_chunk(Arow1, k0 + half * 8, B0, B1, B2, B3, k0 + half * 8,
                  acc0, acc1, acc2, acc3);
    if (Arow2)
        for (int k0 = K1; k0 < K; k0 += 32)
            mma_chunk(Arow2, (k0 - K1) + half * 8, B0, B1, B2, B3, k0 + half * 8,
                      acc0, acc1, acc2, acc3);

    v8f accs[4] = {acc0, acc1, acc2, acc3};
    #pragma unroll
    for (int t = 0; t < 4; ++t) {
        const int nc = n0 + t * 16 + l16;
        const float bv = bias ? bias[nc] : 0.0f;
        #pragma unroll
        for (int v = 0; v < 8; ++v) {
            // C/D layout: lanes 0-15: M = m0+v, N = lane; lanes 16-31: M = m0+v+8.
            const int rr = m0 + v + half * 8;
            float x = accs[t][v] + bv;
            if (ACT == 1) x = tanhf(x);
            if (C)  C[(size_t)rr * N + nc]  = x;
            if (Cb) Cb[(size_t)rr * N + nc] = (bf16_t)x;
        }
    }
}

// ------------------------------ small kernels ------------------------------

__global__ void cvt_bf16(const float* __restrict__ s, bf16_t* __restrict__ d, int n) {
    int i = blockIdx.x * blockDim.x + threadIdx.x;
    if (i < n) d[i] = (bf16_t)s[i];
}

// dst(N x K, bf16) = transpose of src(K x N, f32)
__global__ void cvt_bf16_t(const float* __restrict__ s, bf16_t* __restrict__ d, int K, int N) {
    int i = blockIdx.x * blockDim.x + threadIdx.x;
    if (i < K * N) {
        int k = i / N, n = i % N;
        d[(size_t)n * K + k] = (bf16_t)s[i];
    }
}

// xe[r, j] = x[r,0]*We[0,j] + x[r,1]*We[1,j] + be[j]   (XE = 64)
__global__ void embed2(const float* __restrict__ x, const float* __restrict__ We,
                       const float* __restrict__ be, bf16_t* __restrict__ out, int rows) {
    int i = blockIdx.x * blockDim.x + threadIdx.x;
    if (i >= rows * 64) return;
    int r = i >> 6, j = i & 63;
    float v = x[2 * r] * We[j] + x[2 * r + 1] * We[64 + j] + be[j];
    out[i] = (bf16_t)v;
}

// PyTorch GRUCell pointwise (gate order r,z,n); biases already folded into Gi/Gh.
__global__ void gru_pw(const float* __restrict__ Gi, const float* __restrict__ Gh,
                       const float* __restrict__ h, float* __restrict__ h2,
                       bf16_t* __restrict__ h2b, int Bsz, int H) {
    int i = blockIdx.x * blockDim.x + threadIdx.x;
    if (i >= Bsz * H) return;
    int b = i / H, j = i % H;
    const float* gi = Gi + (size_t)b * 3 * H;
    const float* gh = Gh + (size_t)b * 3 * H;
    float r = sigmoidf_(gi[j] + gh[j]);
    float z = sigmoidf_(gi[H + j] + gh[H + j]);
    float n = tanhf(gi[2 * H + j] + r * gh[2 * H + j]);
    float o = (1.0f - z) * n + z * h[i];
    h2[i] = o;
    h2b[i] = (bf16_t)o;
}

// dest feats (mid, cos, sin) fused with the K=4 Wde matmul -> de (rows x 64, bf16)
__global__ void feats_de(const float* __restrict__ dest, const float* __restrict__ Wde,
                         const float* __restrict__ bde, bf16_t* __restrict__ de, int rows) {
    int i = blockIdx.x * blockDim.x + threadIdx.x;
    if (i >= rows * 64) return;
    int r = i >> 6, j = i & 63;
    const float* d = dest + (size_t)r * 4;
    float mx = 0.5f * (d[0] + d[2]);
    float my = 0.5f * (d[1] + d[3]);
    float rr = sqrtf(mx * mx + my * my);
    float c = 1.0f, s = 0.0f;                 // atan2(0,0)=0 -> cos=1, sin=0
    if (rr > 0.0f) { c = mx / rr; s = my / rr; }
    float v = mx * Wde[j] + my * Wde[64 + j] + c * Wde[128 + j] + s * Wde[192 + j] + bde[j];
    de[i] = (bf16_t)v;
}

// out[r] = dot(A_bf16[r,:K], w_f32[:K]) + bias[0]; one wave per row.
__global__ void dot_rows(const bf16_t* __restrict__ A, const float* __restrict__ w,
                         const float* __restrict__ bias, float* __restrict__ out,
                         int rows, int K) {
    int r = blockIdx.x * (blockDim.x >> 5) + (threadIdx.x >> 5);
    if (r >= rows) return;
    int lane = threadIdx.x & 31;
    const bf16_t* a = A + (size_t)r * K;
    float s = 0.0f;
    for (int c = lane; c < K; c += 32) s += (float)a[c] * w[c];
    for (int off = 16; off; off >>= 1) s += __shfl_xor(s, off, 32);
    if (lane == 0) out[r] = s + bias[0];
}

// scores: softmax over N_DEST per batch row (input strided s[d*B+b]).
__global__ void softmax_scores(const float* __restrict__ s, float* __restrict__ out,
                               int Bsz, int ND) {
    int b = blockIdx.x * blockDim.x + threadIdx.x;
    if (b >= Bsz) return;
    float m = -1e30f;
    for (int d = 0; d < ND; ++d) m = fmaxf(m, s[(size_t)d * Bsz + b]);
    float sum = 0.0f;
    for (int d = 0; d < ND; ++d) sum += __expf(s[(size_t)d * Bsz + b] - m);
    float inv = 1.0f / sum;
    for (int d = 0; d < ND; ++d)
        out[(size_t)b * ND + d] = __expf(s[(size_t)d * Bsz + b] - m) * inv;
}

// Fused decoder attention: one block per batch element b.
// Waves 0..ND-1: comp[d] = dot(C1b[(d*B+b), :256], Wc2) + bc2 -> LDS.
// Then 128 threads: alpha = softmax(comp); w[b,c] = sum_d alpha_d * cat_emb[d*B+b, c].
__global__ __launch_bounds__(640)
void comp_attend(const bf16_t* __restrict__ C1b, const float* __restrict__ Wc2,
                 const float* __restrict__ bc2, const bf16_t* __restrict__ catb,
                 float* __restrict__ w, int Bsz, int ND) {
    __shared__ float comp_s[32];
    const int b = blockIdx.x;
    const int wave = threadIdx.x >> 5, lane = threadIdx.x & 31;
    if (wave < ND) {
        const bf16_t* a = C1b + ((size_t)wave * Bsz + b) * 256;
        float s = 0.0f;
        for (int c = lane; c < 256; c += 32) s += (float)a[c] * Wc2[c];
        for (int off = 16; off; off >>= 1) s += __shfl_xor(s, off, 32);
        if (lane == 0) comp_s[wave] = s + bc2[0];
    }
    __syncthreads();
    const int c = threadIdx.x;
    if (c < 128) {
        float m = -1e30f;
        for (int d = 0; d < ND; ++d) m = fmaxf(m, comp_s[d]);
        float sum = 0.0f, acc = 0.0f;
        for (int d = 0; d < ND; ++d) {
            float e = __expf(comp_s[d] - m);
            sum += e;
            acc += e * (float)catb[((size_t)d * Bsz + b) * 128 + c];
        }
        w[(size_t)b * 128 + c] = acc / sum;
    }
}

// xe2 = [w | y] @ Wxe(130x64) + bxe
__global__ void xe2_kernel(const float* __restrict__ w, const float* __restrict__ y,
                           const float* __restrict__ Wxe, const float* __restrict__ bxe,
                           bf16_t* __restrict__ xe2, int Bsz) {
    int i = blockIdx.x * blockDim.x + threadIdx.x;
    if (i >= Bsz * 64) return;
    int b = i >> 6, j = i & 63;
    float s = bxe[j];
    const float* wr = w + (size_t)b * 128;
    for (int c = 0; c < 128; ++c) s += wr[c] * Wxe[(size_t)c * 64 + j];
    s += y[2 * b] * Wxe[128 * 64 + j] + y[2 * b + 1] * Wxe[129 * 64 + j];
    xe2[i] = (bf16_t)s;
}

// y = h @ Wout(256x2) + bout; also writes the traj slice for this step.
__global__ void out_proj(const float* __restrict__ h, const float* __restrict__ Wout,
                         const float* __restrict__ bout, float* __restrict__ y,
                         float* __restrict__ traj, int Bsz) {
    int i = blockIdx.x * blockDim.x + threadIdx.x;
    if (i >= Bsz * 2) return;
    int b = i >> 1, k = i & 1;
    float s = bout[k];
    const float* hr = h + (size_t)b * 256;
    for (int j = 0; j < 256; ++j) s += hr[j] * Wout[2 * j + k];
    y[i] = s;
    traj[i] = s;
}

// --------------------------------- host ------------------------------------

extern "C" void kernel_launch(void* const* d_in, const int* in_sizes, int n_in,
                              void* d_out, int out_size, void* d_ws, size_t ws_size,
                              hipStream_t stream)
{
    (void)in_sizes; (void)n_in; (void)out_size;
    const int Bsz = 2048, OBS = 8, PRED = 12, ND = 20;
    const int XE = 64, ZO = 256, ZG = 128, CC = 128;
    const int R = ND * Bsz;  // 40960

    const float* x_rel  = (const float*)d_in[0];
    const float* dest   = (const float*)d_in[2];
    const float* We_zo  = (const float*)d_in[3];
    const float* be_zo  = (const float*)d_in[4];
    const float* Wih_zo = (const float*)d_in[5];
    const float* Whh_zo = (const float*)d_in[6];
    const float* bih_zo = (const float*)d_in[7];
    const float* bhh_zo = (const float*)d_in[8];
    const float* We_zg  = (const float*)d_in[9];
    const float* be_zg  = (const float*)d_in[10];
    const float* Wih_zg = (const float*)d_in[11];
    const float* Whh_zg = (const float*)d_in[12];
    const float* bih_zg = (const float*)d_in[13];
    const float* bhh_zg = (const float*)d_in[14];
    const float* Wde    = (const float*)d_in[15];
    const float* bde    = (const float*)d_in[16];
    const float* Wcs    = (const float*)d_in[17];
    const float* bcs    = (const float*)d_in[18];
    const float* Wps    = (const float*)d_in[19];
    const float* bps    = (const float*)d_in[20];
    const float* Wc1    = (const float*)d_in[21];
    const float* bc1    = (const float*)d_in[22];
    const float* Wc2    = (const float*)d_in[23];
    const float* bc2    = (const float*)d_in[24];
    const float* Wxe    = (const float*)d_in[25];
    const float* bxe    = (const float*)d_in[26];
    const float* Wih_d  = (const float*)d_in[27];
    const float* Whh_d  = (const float*)d_in[28];
    const float* bih_d  = (const float*)d_in[29];
    const float* bhh_d  = (const float*)d_in[30];
    const float* Wout   = (const float*)d_in[31];
    const float* bout   = (const float*)d_in[32];

    float* traj_out   = (float*)d_out;                       // (12, 2048, 2)
    float* scores_out = traj_out + (size_t)PRED * Bsz * 2;   // (2048, 20)

    // workspace carve-out (256B aligned)
    char* base = (char*)d_ws;
    size_t used = 0;
    auto alloc = [&](size_t bytes) -> void* {
        bytes = (bytes + 255) & ~(size_t)255;
        void* r = base + used; used += bytes; return r;
    };
    bf16_t* bWih_zo = (bf16_t*)alloc((size_t)3 * ZO * XE * 2);
    bf16_t* bWhh_zo = (bf16_t*)alloc((size_t)3 * ZO * ZO * 2);
    bf16_t* bWih_zg = (bf16_t*)alloc((size_t)3 * ZG * XE * 2);
    bf16_t* bWhh_zg = (bf16_t*)alloc((size_t)3 * ZG * ZG * 2);
    bf16_t* bWih_d  = (bf16_t*)alloc((size_t)3 * ZO * XE * 2);
    bf16_t* bWhh_d  = (bf16_t*)alloc((size_t)3 * ZO * ZO * 2);
    bf16_t* bWcsT   = (bf16_t*)alloc((size_t)128 * 192 * 2);
    bf16_t* bWc1T   = (bf16_t*)alloc((size_t)256 * 384 * 2);
    bf16_t* xeo     = (bf16_t*)alloc((size_t)OBS * Bsz * XE * 2);
    bf16_t* xeg     = (bf16_t*)alloc((size_t)OBS * Bsz * XE * 2);
    float*  h_zo    = (float*) alloc((size_t)Bsz * ZO * 4);
    bf16_t* hb_zo   = (bf16_t*)alloc((size_t)Bsz * ZO * 2);
    float*  h_zg    = (float*) alloc((size_t)Bsz * ZG * 4);
    bf16_t* hb_zg   = (bf16_t*)alloc((size_t)Bsz * ZG * 2);
    float*  Gi      = (float*) alloc((size_t)Bsz * 3 * ZO * 4);
    float*  Gh      = (float*) alloc((size_t)Bsz * 3 * ZO * 4);
    bf16_t* de      = (bf16_t*)alloc((size_t)R * 64 * 2);
    bf16_t* catb    = (bf16_t*)alloc((size_t)R * 128 * 2);
    float*  sc      = (float*) alloc((size_t)R * 4);
    bf16_t* C1b     = (bf16_t*)alloc((size_t)R * 256 * 2);
    float*  wctx    = (float*) alloc((size_t)Bsz * 128 * 4);
    bf16_t* xe2     = (bf16_t*)alloc((size_t)Bsz * XE * 2);
    float*  ybuf    = (float*) alloc((size_t)Bsz * 2 * 4);
    if (used > ws_size) return;  // insufficient scratch: do nothing (deterministic)

    // plain GEMM (single A source)
    auto gemm = [&](int act, const bf16_t* A, int lda, const bf16_t* Bt, int ldb,
                    const float* bias, float* C, bf16_t* Cb, int M, int N, int K) {
        dim3 g(M / 128, N / 64);
        if (act) gemm_wmma_bf16<1><<<g, 256, 0, stream>>>(A, lda, 0, nullptr, 0, 0,
                                                          Bt, ldb, bias, C, Cb, M, N, K, K);
        else     gemm_wmma_bf16<0><<<g, 256, 0, stream>>>(A, lda, 0, nullptr, 0, 0,
                                                          Bt, ldb, bias, C, Cb, M, N, K, K);
    };
    // split-A GEMM: A = [A1 | A2] along K, with per-part row-tiling modulus
    auto gemm2 = [&](int act, const bf16_t* A1, int lda1, int mod1,
                     const bf16_t* A2, int lda2, int mod2,
                     const bf16_t* Bt, int ldb, const float* bias,
                     float* C, bf16_t* Cb, int M, int N, int K1, int K) {
        dim3 g(M / 128, N / 64);
        if (act) gemm_wmma_bf16<1><<<g, 256, 0, stream>>>(A1, lda1, mod1, A2, lda2, mod2,
                                                          Bt, ldb, bias, C, Cb, M, N, K1, K);
        else     gemm_wmma_bf16<0><<<g, 256, 0, stream>>>(A1, lda1, mod1, A2, lda2, mod2,
                                                          Bt, ldb, bias, C, Cb, M, N, K1, K);
    };
    auto cvt = [&](const float* s, bf16_t* d, int n) {
        cvt_bf16<<<(n + 255) / 256, 256, 0, stream>>>(s, d, n);
    };

    // ---- weight prep (bf16, N-major) ----
    cvt(Wih_zo, bWih_zo, 3 * ZO * XE);
    cvt(Whh_zo, bWhh_zo, 3 * ZO * ZO);
    cvt(Wih_zg, bWih_zg, 3 * ZG * XE);
    cvt(Whh_zg, bWhh_zg, 3 * ZG * ZG);
    cvt(Wih_d,  bWih_d,  3 * ZO * XE);
    cvt(Whh_d,  bWhh_d,  3 * ZO * ZO);
    cvt_bf16_t<<<(192 * 128 + 255) / 256, 256, 0, stream>>>(Wcs, bWcsT, 192, 128);
    cvt_bf16_t<<<(384 * 256 + 255) / 256, 256, 0, stream>>>(Wc1, bWc1T, 384, 256);

    // ---- input embeddings (K=2, scalar) ----
    embed2<<<(OBS * Bsz * 64 + 255) / 256, 256, 0, stream>>>(x_rel, We_zo, be_zo, xeo, OBS * Bsz);
    embed2<<<(OBS * Bsz * 64 + 255) / 256, 256, 0, stream>>>(x_rel, We_zg, be_zg, xeg, OBS * Bsz);

    // ---- encoders (sequential GRU scans) ----
    hipMemsetAsync(h_zo,  0, (size_t)Bsz * ZO * 4, stream);
    hipMemsetAsync(hb_zo, 0, (size_t)Bsz * ZO * 2, stream);
    hipMemsetAsync(h_zg,  0, (size_t)Bsz * ZG * 4, stream);
    hipMemsetAsync(hb_zg, 0, (size_t)Bsz * ZG * 2, stream);
    for (int t = 0; t < OBS; ++t) {
        gemm(0, xeo + (size_t)t * Bsz * XE, XE, bWih_zo, XE, bih_zo, Gi, nullptr, Bsz, 3 * ZO, XE);
        gemm(0, hb_zo, ZO, bWhh_zo, ZO, bhh_zo, Gh, nullptr, Bsz, 3 * ZO, ZO);
        gru_pw<<<(Bsz * ZO + 255) / 256, 256, 0, stream>>>(Gi, Gh, h_zo, h_zo, hb_zo, Bsz, ZO);
    }
    for (int t = 0; t < OBS; ++t) {
        gemm(0, xeg + (size_t)t * Bsz * XE, XE, bWih_zg, XE, bih_zg, Gi, nullptr, Bsz, 3 * ZG, XE);
        gemm(0, hb_zg, ZG, bWhh_zg, ZG, bhh_zg, Gh, nullptr, Bsz, 3 * ZG, ZG);
        gru_pw<<<(Bsz * ZG + 255) / 256, 256, 0, stream>>>(Gi, Gh, h_zg, h_zg, hb_zg, Bsz, ZG);
    }

    // ---- score network ----
    feats_de<<<(R * 64 + 255) / 256, 256, 0, stream>>>(dest, Wde, bde, de, R);
    // cat_emb = tanh([tile(zg) | de] @ Wcs + bcs); zg read in place via row % B
    gemm2(1, hb_zg, ZG, Bsz, de, 64, 0, bWcsT, 192, bcs,
          nullptr, catb, R, CC, ZG, ZG + 64);
    dot_rows<<<(R + 7) / 8, 256, 0, stream>>>(catb, Wps, bps, sc, R, CC);
    softmax_scores<<<(Bsz + 255) / 256, 256, 0, stream>>>(sc, scores_out, Bsz, ND);

    // ---- decoder: carry (h = ho, y = x_rel[-1]) ----
    hipMemcpyAsync(ybuf, x_rel + (size_t)(OBS - 1) * Bsz * 2,
                   (size_t)Bsz * 2 * sizeof(float), hipMemcpyDeviceToDevice, stream);
    for (int t = 0; t < PRED; ++t) {
        // tanh([cat_emb | tile(h)] @ Wc1 + bc1); h read in place via row % B
        gemm2(1, catb, CC, 0, hb_zo, ZO, Bsz, bWc1T, 384, bc1,
              nullptr, C1b, R, ZO, CC, CC + ZO);
        comp_attend<<<Bsz, 640, 0, stream>>>(C1b, Wc2, bc2, catb, wctx, Bsz, ND);
        xe2_kernel<<<(Bsz * 64 + 255) / 256, 256, 0, stream>>>(wctx, ybuf, Wxe, bxe, xe2, Bsz);
        gemm(0, xe2, XE, bWih_d, XE, bih_d, Gi, nullptr, Bsz, 3 * ZO, XE);
        gemm(0, hb_zo, ZO, bWhh_d, ZO, bhh_d, Gh, nullptr, Bsz, 3 * ZO, ZO);
        gru_pw<<<(Bsz * ZO + 255) / 256, 256, 0, stream>>>(Gi, Gh, h_zo, h_zo, hb_zo, Bsz, ZO);
        out_proj<<<(Bsz * 2 + 255) / 256, 256, 0, stream>>>(h_zo, Wout, bout, ybuf,
                                                            traj_out + (size_t)t * Bsz * 2, Bsz);
    }
}
